// GAT_18889266168312
// MI455X (gfx1250) — compile-verified, matrix operand order checked by reading, not run on hardware
//
#include <hip/hip_runtime.h>
#include <hip/hip_bf16.h>
#include <math.h>

typedef __attribute__((ext_vector_type(16))) _Float16 v16h;
typedef __attribute__((ext_vector_type(8)))  float    v8f;
typedef __attribute__((ext_vector_type(4)))  int      int32x4;

#define HEADS   4
#define OUT_CH  32
#define HC      128   // HEADS*OUT_CH
#define IN_DIM  128
#define NEG_SLOPE 0.2f
#define KSTEPS  (IN_DIM / 32)          // 4
#define NTILES  (HC / 16)              // 8
// packed W: [tileN][kstep][lane][16 halfs] = 8*4*32*16 halfs = 32 KB
#define BPACK_HALFS (NTILES * KSTEPS * 32 * 16)

// Async global->LDS copy path (CDNA5 GLOBAL_LOAD_ASYNC_TO_LDS_B128, ASYNCcnt),
// gated on builtin availability so compilation can never regress.
#if defined(__has_builtin)
# if __has_builtin(__builtin_amdgcn_global_load_async_to_lds_b128) && \
     __has_builtin(__builtin_amdgcn_s_wait_asynccnt)
#  define USE_ASYNC_LDS 1
# endif
#endif

// pointer types matching the builtin's parameters:
//   arg0: int32x4 addrspace(1)*  (global, printed by clang as "__device__")
//   arg1: int32x4 addrspace(3)*  (LDS)
typedef __attribute__((address_space(1))) int32x4* gp128_t;
typedef __attribute__((address_space(3))) int32x4* lp128_t;

// ---------------------------------------------------------------------------
// Repack W (128x128 f32, row-major) into per-lane WMMA B-fragment order (f16).
// Fragment element j (0..15): vgpr=j/2, p=j&1;
//   vgpr<4 : k_local = hi*8 + 2*vgpr + p
//   vgpr>=4: k_local = 16 + hi*8 + 2*(vgpr-4) + p     (hi = lane>>4)
// n = tileN*16 + (lane&15);  k = kstep*32 + k_local
// ---------------------------------------------------------------------------
__global__ void gat_pack_W(const float* __restrict__ W,
                           _Float16* __restrict__ Bpack) {
    int t = blockIdx.x * blockDim.x + threadIdx.x;      // 0 .. 1023
    if (t >= NTILES * KSTEPS * 32) return;
    int lane = t & 31;
    int kki  = (t >> 5) & (KSTEPS - 1);
    int tn   = t >> 7;
    int hi   = lane >> 4;
    int n    = tn * 16 + (lane & 15);
#pragma unroll
    for (int j = 0; j < 16; j++) {
        int vg = j >> 1, p = j & 1;
        int klocal = (vg < 4) ? (hi * 8 + 2 * vg + p)
                              : (16 + hi * 8 + 2 * (vg - 4) + p);
        int k = kki * 32 + klocal;
        Bpack[t * 16 + j] = (_Float16)W[k * HC + n];
    }
}

// ---------------------------------------------------------------------------
// WMMA GEMM: feat (Ntot x 128, f32) @ W (packed f16) -> Xout (Ntot x 128, f32)
// Block = 8 waves; block b handles M-tile b, wave w handles N-tile w.
// Packed W staged in LDS once per block (async copy when available);
// A converted f32->f16 in registers.
// ---------------------------------------------------------------------------
__global__ void gat_gemm_wmma(const float* __restrict__ A,       // Ntot x 128
                              const _Float16* __restrict__ Bpack,
                              float* __restrict__ C,             // Ntot x 128
                              int M, int tilesM) {
    __shared__ _Float16 ldsB[BPACK_HALFS];                       // 32 KB

#ifdef USE_ASYNC_LDS
    {
        char* gbase = (char*)Bpack;
        char* lbase = (char*)ldsB;
        // 2048 x 16B chunks; each lane moves one b128 per issue (512B/wave)
        for (int i = threadIdx.x; i < BPACK_HALFS / 8; i += blockDim.x) {
            __builtin_amdgcn_global_load_async_to_lds_b128(
                (gp128_t)(gbase + (size_t)i * 16),
                (lp128_t)(lbase + (size_t)i * 16),
                /*offset=*/0, /*cpol=*/0);
        }
        __builtin_amdgcn_s_wait_asynccnt(0);
    }
#else
    {
        const uint4* gsrc = (const uint4*)Bpack;
        uint4* ldst = (uint4*)ldsB;
        const int n16 = BPACK_HALFS / 8;                         // 2048 uint4
        for (int i = threadIdx.x; i < n16; i += blockDim.x) ldst[i] = gsrc[i];
    }
#endif
    __syncthreads();

    const int wave = threadIdx.x >> 5;                           // = tileN
    const int lane = threadIdx.x & 31;
    const int tileM = blockIdx.x;
    if (tileM >= tilesM) return;                                 // block-uniform

    const int row = lane & 15;                                   // M-row / N-col
    const int hi  = lane >> 4;

    int arow = tileM * 16 + row;
    if (arow >= M) arow = M - 1;
    const float* Arow = A + (size_t)arow * IN_DIM;

    v8f acc = {};
#pragma unroll
    for (int kki = 0; kki < KSTEPS; kki++) {
        const int kk = kki * 32;
        if (kk + 32 < IN_DIM)
            __builtin_prefetch(Arow + kk + 32, 0, 1);            // global_prefetch_b8

        // ---- A fragment: two contiguous 8-float chunks, cvt to f16 --------
        const float* Ar = Arow + kk + hi * 8;
        float4 c0 = *(const float4*)(Ar + 0);
        float4 c1 = *(const float4*)(Ar + 4);
        float4 c2 = *(const float4*)(Ar + 16);
        float4 c3 = *(const float4*)(Ar + 20);
        v16h a;
        a[0] = (_Float16)c0.x;  a[1] = (_Float16)c0.y;
        a[2] = (_Float16)c0.z;  a[3] = (_Float16)c0.w;
        a[4] = (_Float16)c1.x;  a[5] = (_Float16)c1.y;
        a[6] = (_Float16)c1.z;  a[7] = (_Float16)c1.w;
        a[8]  = (_Float16)c2.x; a[9]  = (_Float16)c2.y;
        a[10] = (_Float16)c2.z; a[11] = (_Float16)c2.w;
        a[12] = (_Float16)c3.x; a[13] = (_Float16)c3.y;
        a[14] = (_Float16)c3.z; a[15] = (_Float16)c3.w;

        // ---- B fragment: 32 contiguous bytes from LDS ---------------------
        const v16h b = *(const v16h*)&ldsB[((wave * KSTEPS + kki) * 32 + lane) * 16];

        acc = __builtin_amdgcn_wmma_f32_16x16x32_f16(
            /*neg_a=*/false, a, /*neg_b=*/false, b,
            /*c_mod=*/(short)0, acc, /*reuse_a=*/false, /*reuse_b=*/false);
    }

    // ---- C store: VGPR j -> M = tileM*16 + j + 8*hi, N = wave*16 + row ----
    float* Cc = C + wave * 16 + row;
#pragma unroll
    for (int j = 0; j < 8; j++) {
        int r = tileM * 16 + j + 8 * hi;
        if (r < M) Cc[(size_t)r * HC] = acc[j];
    }
}

// ---------------------------------------------------------------------------
// Per-(node, head) attention logits
// ---------------------------------------------------------------------------
__global__ void gat_logits(const float* __restrict__ x,
                           const float* __restrict__ att_src,
                           const float* __restrict__ att_dst,
                           float* __restrict__ a_src,
                           float* __restrict__ a_dst, int Ntot) {
    int t = blockIdx.x * blockDim.x + threadIdx.x;
    if (t >= Ntot * HEADS) return;
    int n = t >> 2, h = t & 3;
    const float* xr = x + (size_t)n * HC + h * OUT_CH;
    float ss = 0.f, sd = 0.f;
#pragma unroll
    for (int oc = 0; oc < OUT_CH; oc++) {
        ss = fmaf(xr[oc], att_src[h * OUT_CH + oc], ss);
        sd = fmaf(xr[oc], att_dst[h * OUT_CH + oc], sd);
    }
    a_src[t] = ss;
    a_dst[t] = sd;
}

// ---------------------------------------------------------------------------
// init: e_max = -inf, denom = 0, out = 0
// ---------------------------------------------------------------------------
__global__ void gat_init(float* __restrict__ emax, float* __restrict__ denom,
                         float* __restrict__ out, int Ntot) {
    int t = blockIdx.x * blockDim.x + threadIdx.x;
    int stride = gridDim.x * blockDim.x;
    int nh = Ntot * HEADS;
    for (int i = t; i < nh; i += stride) { emax[i] = -__builtin_inff(); denom[i] = 0.f; }
    int no = Ntot * HC;
    for (int i = t; i < no; i += stride) out[i] = 0.f;
}

// order-preserving float atomic max (exact; buffer initialized to -inf)
__device__ inline void atomicMaxFloat(float* addr, float v) {
    if (v >= 0.f) atomicMax((int*)addr, __float_as_int(v));
    else          atomicMin((unsigned int*)addr, __float_as_uint(v));
}

__device__ inline void edge_ends(const int* __restrict__ src,
                                 const int* __restrict__ dst,
                                 int e, int E, int& s, int& d) {
    if (e < E) { s = src[e]; d = dst[e]; }
    else       { s = d = e - E; }               // self loop
}

// ---------------------------------------------------------------------------
// pass 1: segment max of leaky-relu logits over dst
// ---------------------------------------------------------------------------
__global__ void gat_edge_max(const float* __restrict__ a_src,
                             const float* __restrict__ a_dst,
                             const int* __restrict__ src,
                             const int* __restrict__ dst,
                             float* __restrict__ emax, int E, int Ntot) {
    int e = blockIdx.x * blockDim.x + threadIdx.x;
    if (e >= E + Ntot) return;
    int s, d; edge_ends(src, dst, e, E, s, d);
#pragma unroll
    for (int h = 0; h < HEADS; h++) {
        float v = a_src[s * HEADS + h] + a_dst[d * HEADS + h];
        v = v > 0.f ? v : NEG_SLOPE * v;
        atomicMaxFloat(&emax[d * HEADS + h], v);
    }
}

// ---------------------------------------------------------------------------
// pass 2: denom = segment sum of exp(e - max)
// ---------------------------------------------------------------------------
__global__ void gat_edge_sum(const float* __restrict__ a_src,
                             const float* __restrict__ a_dst,
                             const int* __restrict__ src,
                             const int* __restrict__ dst,
                             const float* __restrict__ emax,
                             float* __restrict__ denom, int E, int Ntot) {
    int e = blockIdx.x * blockDim.x + threadIdx.x;
    if (e >= E + Ntot) return;
    int s, d; edge_ends(src, dst, e, E, s, d);
#pragma unroll
    for (int h = 0; h < HEADS; h++) {
        float v = a_src[s * HEADS + h] + a_dst[d * HEADS + h];
        v = v > 0.f ? v : NEG_SLOPE * v;
        atomicAdd(&denom[d * HEADS + h], expf(v - emax[d * HEADS + h]));
    }
}

// ---------------------------------------------------------------------------
// pass 3: out[dst] += x[src] * alpha     (per (edge, head) thread)
// ---------------------------------------------------------------------------
__global__ void gat_edge_aggr(const float* __restrict__ a_src,
                              const float* __restrict__ a_dst,
                              const int* __restrict__ src,
                              const int* __restrict__ dst,
                              const float* __restrict__ emax,
                              const float* __restrict__ denom,
                              const float* __restrict__ x,
                              float* __restrict__ out, int E, int Ntot) {
    int t = blockIdx.x * blockDim.x + threadIdx.x;
    if (t >= (E + Ntot) * HEADS) return;
    int e = t >> 2, h = t & 3;
    int s, d; edge_ends(src, dst, e, E, s, d);
    float v = a_src[s * HEADS + h] + a_dst[d * HEADS + h];
    v = v > 0.f ? v : NEG_SLOPE * v;
    float p = expf(v - emax[d * HEADS + h]);
    float alpha = p / (denom[d * HEADS + h] + 1e-16f);
    const float* xs = x + (size_t)s * HC + h * OUT_CH;
    float* od = out + (size_t)d * HC + h * OUT_CH;
#pragma unroll
    for (int oc = 0; oc < OUT_CH; oc += 4) {
        float4 m = *(const float4*)(xs + oc);
        atomicAdd(&od[oc + 0], m.x * alpha);
        atomicAdd(&od[oc + 1], m.y * alpha);
        atomicAdd(&od[oc + 2], m.z * alpha);
        atomicAdd(&od[oc + 3], m.w * alpha);
    }
}

// ---------------------------------------------------------------------------
// bias add (bias is zero in reference, applied faithfully)
// ---------------------------------------------------------------------------
__global__ void gat_bias(float* __restrict__ out,
                         const float* __restrict__ bias, int Ntot) {
    int t = blockIdx.x * blockDim.x + threadIdx.x;
    int stride = gridDim.x * blockDim.x;
    int no = Ntot * HC;
    for (int i = t; i < no; i += stride) out[i] += bias[i & (HC - 1)];
}

// ---------------------------------------------------------------------------
extern "C" void kernel_launch(void* const* d_in, const int* in_sizes, int n_in,
                              void* d_out, int out_size, void* d_ws, size_t ws_size,
                              hipStream_t stream) {
    const float* feat    = (const float*)d_in[0];   // (B,N,128)
    const int*   eidx    = (const int*)  d_in[1];   // (2, E)
    const float* W       = (const float*)d_in[2];   // (128, 128)
    const float* att_src = (const float*)d_in[3];   // (4, 32)
    const float* att_dst = (const float*)d_in[4];   // (4, 32)
    const float* bias    = (const float*)d_in[5];   // (128,)
    float* out = (float*)d_out;

    const int Ntot = in_sizes[0] / IN_DIM;          // 40000
    const int E    = in_sizes[1] / 2;               // 640000
    const int* src = eidx;
    const int* dst = eidx + E;

    // workspace carving (256B aligned)
    auto align = [](size_t v) { return (v + 255) & ~(size_t)255; };
    char* ws = (char*)d_ws;
    size_t off = 0;
    _Float16* Bpack = (_Float16*)(ws + off); off = align(off + (size_t)BPACK_HALFS * 2);
    float*    xf    = (float*)   (ws + off); off = align(off + (size_t)Ntot * HC * 4);
    float*    asrc  = (float*)   (ws + off); off = align(off + (size_t)Ntot * HEADS * 4);
    float*    adst  = (float*)   (ws + off); off = align(off + (size_t)Ntot * HEADS * 4);
    float*    emax  = (float*)   (ws + off); off = align(off + (size_t)Ntot * HEADS * 4);
    float*    denom = (float*)   (ws + off); off = align(off + (size_t)Ntot * HEADS * 4);
    (void)ws_size; (void)n_in; (void)out_size;

    const int T = 256;

    // 1. repack W into WMMA fragment order (f16)
    gat_pack_W<<<(NTILES * KSTEPS * 32 + T - 1) / T, T, 0, stream>>>(W, Bpack);

    // 2. WMMA GEMM: block per M-tile, wave per N-tile, W staged in LDS
    int tilesM = (Ntot + 15) / 16;
    gat_gemm_wmma<<<tilesM, T, 0, stream>>>(feat, Bpack, xf, Ntot, tilesM);

    // 3. logits
    gat_logits<<<(Ntot * HEADS + T - 1) / T, T, 0, stream>>>(xf, att_src, att_dst,
                                                             asrc, adst, Ntot);
    // 4. init accumulators + output
    gat_init<<<(Ntot * HC + T - 1) / T, T, 0, stream>>>(emax, denom, out, Ntot);

    // 5-7. edge passes
    int Etot = E + Ntot;
    gat_edge_max <<<(Etot + T - 1) / T, T, 0, stream>>>(asrc, adst, src, dst, emax, E, Ntot);
    gat_edge_sum <<<(Etot + T - 1) / T, T, 0, stream>>>(asrc, adst, src, dst, emax, denom, E, Ntot);
    gat_edge_aggr<<<(Etot * HEADS + T - 1) / T, T, 0, stream>>>(asrc, adst, src, dst,
                                                                emax, denom, xf, out, E, Ntot);
    // 8. bias
    gat_bias<<<(Ntot * HC + T - 1) / T, T, 0, stream>>>(out, bias, Ntot);
}
